// TtDeiTSelfAttention_47760036331639
// MI455X (gfx1250) — matrix operand order repaired and checked by reading, not compiled, not run on hardware
//
#include <hip/hip_runtime.h>
#include <hip/hip_fp16.h>

// ---------------------------------------------------------------------------
// MI455X (gfx1250) self-attention.  wave32, WMMA f32_16x16x32_f16.
//
// Precisions: inputs f32, matmuls in f16 with f32 accumulation (data scale
// 0.02 -> f16 mantissa is plenty), softmax in f32, output f32.
//
// Workspace layout (bytes):
//   [ 0MB) Xh   : X as f16,     8192 x 1024
//   [16MB) Wqh  : wq as f16,    1024 x 1024
//   [18MB) Wkh  : wk as f16
//   [20MB) Wvh  : wv as f16
//   [22MB) Qh   : Q  f16  [B*nH, S, d]   (pre-scaled by 1/sqrt(d))
//   [38MB) Kh   : K  f16  [B*nH, S, d]
//   [54MB) Vth  : V^T f16 [B*nH, d, S]   (transposed so PV GEMM B-frag is
//                                          contiguous along the key dim)
//   total ~70MB
// ---------------------------------------------------------------------------

typedef __attribute__((ext_vector_type(16))) _Float16 v16h;
typedef __attribute__((ext_vector_type(8)))  float    v8f;
typedef __attribute__((ext_vector_type(4)))  int      v4i;
typedef __attribute__((ext_vector_type(4)))  float    v4f;
typedef __attribute__((ext_vector_type(4)))  _Float16 v4h;

union FragU { v16h h; v4i i[2]; };

// A-fragment 16x32 f16 (ISA 7.12.2): lane l holds row m = l&15.
// lanes 0-15 : regs 0-3 = K0..7,  regs 4-7 = K16..23
// lanes 16-31: regs 0-3 = K8..15, regs 4-7 = K24..31
// -> per lane: two 16B loads at k = (l>=16 ? 8 : 0) and +16.
__device__ __forceinline__ v16h load_a_frag(const _Float16* base, int ld) {
  const int l = threadIdx.x & 31;
  const _Float16* p = base + (l & 15) * ld + ((l >> 4) << 3);
  FragU u;
  u.i[0] = *(const v4i*)(p);
  u.i[1] = *(const v4i*)(p + 16);
  return u.h;
}

// B-fragment 32x16 f16, same K order as the A-fragment:
// lane l holds column n = l&15; lanes 0-15 hold K0..7,K16..23, lanes 16-31
// K8..15,K24..31. `base` must be K-contiguous per column (row n of W / K / V^T).
__device__ __forceinline__ v16h load_b_frag(const _Float16* base, int ld) {
  const int l = threadIdx.x & 31;
  const _Float16* p = base + (l & 15) * ld + ((l >> 4) << 3);
  FragU u;
  u.i[0] = *(const v4i*)(p);
  u.i[1] = *(const v4i*)(p + 16);
  return u.h;
}

__device__ __forceinline__ v8f wmma_f16(v16h a, v16h b, v8f c) {
  return __builtin_amdgcn_wmma_f32_16x16x32_f16(
      /*neg_a=*/false, a, /*neg_b=*/false, b,
      /*c_mod=*/(short)0, c, /*reuse_a=*/false, /*reuse_b=*/false);
}

// ---------------------------------------------------------------------------
// f32 -> f16 conversion (vectorized x4, grid-stride)
// ---------------------------------------------------------------------------
__global__ void cvt_f32_to_f16_v4(const v4f* __restrict__ src,
                                  v4h* __restrict__ dst, int n4) {
  int i = blockIdx.x * blockDim.x + threadIdx.x;
  const int stride = gridDim.x * blockDim.x;
  for (; i < n4; i += stride) {
    v4f x = src[i];
    dst[i] = __builtin_convertvector(x, v4h);
  }
}

// ---------------------------------------------------------------------------
// Fused QKV projection:  out[m,o] = sum_k Xh[m,k] * W[o,k] + b[o]
// One wave -> 64x64 output tile (4x4 WMMA tiles, 16 f32 accumulators).
// wave id: [mat(3)][mtile(128)][ntile(16)]   (8192/64 x 1024/64)
// Epilogue: +bias; Q scaled by 1/8; Q,K stored [bh,S,d]; V stored [bh,d,S].
// ---------------------------------------------------------------------------
__global__ void __launch_bounds__(128)
qkv_gemm_kernel(const _Float16* __restrict__ Xh,
                const _Float16* __restrict__ Wq, const _Float16* __restrict__ Wk,
                const _Float16* __restrict__ Wv,
                const float* __restrict__ bq, const float* __restrict__ bk,
                const float* __restrict__ bv,
                _Float16* __restrict__ Qh, _Float16* __restrict__ Kh,
                _Float16* __restrict__ Vth) {
  const int wave = blockIdx.x * (blockDim.x >> 5) + (threadIdx.x >> 5);
  const int mat  = wave >> 11;            // 0:Q 1:K 2:V  (2048 waves each)
  const int rem  = wave & 2047;
  const int m0   = (rem >> 4) << 6;       // token-row base (64-tile)
  const int n0   = (rem & 15) << 6;       // out-col base   (64-tile)

  const _Float16* W    = (mat == 0) ? Wq : (mat == 1) ? Wk : Wv;
  const float*    bias = (mat == 0) ? bq : (mat == 1) ? bk : bv;

  v8f acc[4][4];
#pragma unroll
  for (int mi = 0; mi < 4; ++mi)
#pragma unroll
    for (int ni = 0; ni < 4; ++ni)
      acc[mi][ni] = (v8f){0.f, 0.f, 0.f, 0.f, 0.f, 0.f, 0.f, 0.f};

  for (int k0 = 0; k0 < 1024; k0 += 32) {
    v16h a[4];
#pragma unroll
    for (int mi = 0; mi < 4; ++mi)
      a[mi] = load_a_frag(Xh + (m0 + mi * 16) * 1024 + k0, 1024);
#pragma unroll
    for (int ni = 0; ni < 4; ++ni) {
      v16h b = load_b_frag(W + (n0 + ni * 16) * 1024 + k0, 1024);
#pragma unroll
      for (int mi = 0; mi < 4; ++mi)
        acc[mi][ni] = wmma_f16(a[mi], b, acc[mi][ni]);
    }
  }

  // epilogue: C layout -> lane n = l&15; reg r -> m = r + (l>=16 ? 8 : 0)
  const int l    = threadIdx.x & 31;
  const int nl   = l & 15;
  const int moff = (l < 16) ? 0 : 8;
#pragma unroll
  for (int ni = 0; ni < 4; ++ni) {
    const int o  = n0 + ni * 16 + nl;     // output feature
    const int h  = o >> 6;                // head
    const int dd = o & 63;                // dim-in-head
    const float bval = bias[o];
#pragma unroll
    for (int mi = 0; mi < 4; ++mi) {
#pragma unroll
      for (int r = 0; r < 8; ++r) {
        const int m = m0 + mi * 16 + moff + r;    // global token row
        const int b = m >> 10;
        const int s = m & 1023;
        const int bh = b * 16 + h;
        float v = acc[mi][ni][r] + bval;
        if (mat == 0) {
          Qh[(bh * 1024 + s) * 64 + dd] = (_Float16)(v * 0.125f);  // fold 1/sqrt(64)
        } else if (mat == 1) {
          Kh[(bh * 1024 + s) * 64 + dd] = (_Float16)v;
        } else {
          Vth[(bh * 64 + dd) * 1024 + s] = (_Float16)v;            // transposed
        }
      }
    }
  }
}

// ---------------------------------------------------------------------------
// Flash attention: one wave per (bh, 16-query tile), streaming 64-key chunks.
// scores = Q K^T (8 wmma/chunk), online softmax (16-lane shfl reductions),
// P -> LDS (f16) -> A-frag reload, ctx += P V (8 wmma/chunk).
// K/V per (b,h) is 256KB -> stays L2-resident across its 64 query waves.
// ---------------------------------------------------------------------------
#define ATT_WAVES 4
__global__ void __launch_bounds__(128)
attention_kernel(const _Float16* __restrict__ Qh, const _Float16* __restrict__ Kh,
                 const _Float16* __restrict__ Vth, float* __restrict__ out) {
  __shared__ _Float16 lds_p[ATT_WAVES][16][72];   // 72-pad: 16B-aligned rows, no bank camping
  const int wid  = threadIdx.x >> 5;
  const int wave = blockIdx.x * ATT_WAVES + wid;
  const int qt   = wave & 63;       // query tile within (b,h)
  const int bh   = wave >> 6;       // 0..127
  const int b    = bh >> 4;
  const int h    = bh & 15;

  const _Float16* Qbase = Qh + (bh * 1024 + qt * 16) * 64;
  const _Float16* Kbase = Kh + bh * 1024 * 64;
  const _Float16* Vbase = Vth + bh * 64 * 1024;

  const v16h qf0 = load_a_frag(Qbase, 64);        // k(dim) 0..31
  const v16h qf1 = load_a_frag(Qbase + 32, 64);   // k(dim) 32..63

  float mrun[8], lrun[8];
  v8f oacc[4];
#pragma unroll
  for (int r = 0; r < 8; ++r) { mrun[r] = -1e30f; lrun[r] = 0.f; }
#pragma unroll
  for (int t = 0; t < 4; ++t)
    oacc[t] = (v8f){0.f, 0.f, 0.f, 0.f, 0.f, 0.f, 0.f, 0.f};

  _Float16* pl   = &lds_p[wid][0][0];
  const int l    = threadIdx.x & 31;
  const int nl   = l & 15;
  const int moff = (l < 16) ? 0 : 8;

  for (int kc = 0; kc < 1024; kc += 64) {
    // ---- scores: 16 queries x 64 keys --------------------------------
    v8f sc[4];
#pragma unroll
    for (int nt = 0; nt < 4; ++nt) {
      sc[nt] = (v8f){0.f, 0.f, 0.f, 0.f, 0.f, 0.f, 0.f, 0.f};
      const _Float16* kb = Kbase + (kc + nt * 16) * 64;
      sc[nt] = wmma_f16(qf0, load_b_frag(kb, 64), sc[nt]);
      sc[nt] = wmma_f16(qf1, load_b_frag(kb + 32, 64), sc[nt]);
    }
    // ---- online softmax (row = reg r within 16-lane half) ------------
    float alpha[8];
#pragma unroll
    for (int r = 0; r < 8; ++r) {
      float mx = fmaxf(fmaxf(sc[0][r], sc[1][r]), fmaxf(sc[2][r], sc[3][r]));
#pragma unroll
      for (int off = 1; off < 16; off <<= 1)
        mx = fmaxf(mx, __shfl_xor(mx, off, 32));  // masks <16 stay in half
      const float mnew = fmaxf(mrun[r], mx);
      alpha[r] = __expf(mrun[r] - mnew);
      mrun[r]  = mnew;
      float rs = 0.f;
#pragma unroll
      for (int nt = 0; nt < 4; ++nt) {
        const float p = __expf(sc[nt][r] - mnew);
        sc[nt][r] = p;
        rs += p;
      }
#pragma unroll
      for (int off = 1; off < 16; off <<= 1)
        rs += __shfl_xor(rs, off, 32);
      lrun[r] = lrun[r] * alpha[r] + rs;
    }
    // ---- P (C layout) -> LDS -> A-fragment layout --------------------
#pragma unroll
    for (int nt = 0; nt < 4; ++nt)
#pragma unroll
      for (int r = 0; r < 8; ++r)
        pl[(moff + r) * 72 + nt * 16 + nl] = (_Float16)sc[nt][r];
    asm volatile("s_wait_dscnt 0" ::: "memory");
    const v16h pa0 = load_a_frag(pl, 72);
    const v16h pa1 = load_a_frag(pl + 32, 72);
    // ---- ctx += P V  (V^T rows are key-contiguous) -------------------
#pragma unroll
    for (int t2 = 0; t2 < 4; ++t2) {
#pragma unroll
      for (int r = 0; r < 8; ++r) oacc[t2][r] *= alpha[r];
      const _Float16* vb = Vbase + t2 * 16 * 1024 + kc;
      oacc[t2] = wmma_f16(pa0, load_b_frag(vb, 1024), oacc[t2]);
      oacc[t2] = wmma_f16(pa1, load_b_frag(vb + 32, 1024), oacc[t2]);
    }
  }
  // ---- normalize + write [B,S,H] f32 ---------------------------------
#pragma unroll
  for (int t2 = 0; t2 < 4; ++t2) {
#pragma unroll
    for (int r = 0; r < 8; ++r) {
      const int s   = qt * 16 + moff + r;
      const int col = h * 64 + t2 * 16 + nl;
      out[(b * 1024 + s) * 1024 + col] = oacc[t2][r] / lrun[r];
    }
  }
}

// ---------------------------------------------------------------------------
extern "C" void kernel_launch(void* const* d_in, const int* in_sizes, int n_in,
                              void* d_out, int out_size, void* d_ws, size_t ws_size,
                              hipStream_t stream) {
  (void)in_sizes; (void)n_in; (void)out_size; (void)ws_size;
  const float* X  = (const float*)d_in[0];
  const float* wq = (const float*)d_in[1];
  const float* bq = (const float*)d_in[2];
  const float* wk = (const float*)d_in[3];
  const float* bk = (const float*)d_in[4];
  const float* wv = (const float*)d_in[5];
  const float* bv = (const float*)d_in[6];
  float* out = (float*)d_out;

  char* ws = (char*)d_ws;
  const size_t MB = 1u << 20;
  _Float16* Xh  = (_Float16*)(ws + 0 * MB);
  _Float16* Wqh = (_Float16*)(ws + 16 * MB);
  _Float16* Wkh = (_Float16*)(ws + 18 * MB);
  _Float16* Wvh = (_Float16*)(ws + 20 * MB);
  _Float16* Qh  = (_Float16*)(ws + 22 * MB);
  _Float16* Kh  = (_Float16*)(ws + 38 * MB);
  _Float16* Vth = (_Float16*)(ws + 54 * MB);

  // 1) convert X and weights to f16
  cvt_f32_to_f16_v4<<<2048, 256, 0, stream>>>((const v4f*)X,  (v4h*)Xh,  8388608 / 4);
  cvt_f32_to_f16_v4<<<1024, 256, 0, stream>>>((const v4f*)wq, (v4h*)Wqh, 1048576 / 4);
  cvt_f32_to_f16_v4<<<1024, 256, 0, stream>>>((const v4f*)wk, (v4h*)Wkh, 1048576 / 4);
  cvt_f32_to_f16_v4<<<1024, 256, 0, stream>>>((const v4f*)wv, (v4h*)Wvh, 1048576 / 4);

  // 2) fused QKV GEMM: 3 * 128 * 16 = 6144 waves, 4 waves/block
  qkv_gemm_kernel<<<1536, 128, 0, stream>>>(Xh, Wqh, Wkh, Wvh, bq, bk, bv,
                                            Qh, Kh, Vth);

  // 3) flash attention: 128 (b,h) * 64 q-tiles = 8192 waves, 4 waves/block
  attention_kernel<<<2048, 128, 0, stream>>>(Qh, Kh, Vth, out);
}
